// FusedExpertsModule_39204461478556
// MI455X (gfx1250) — matrix operand order repaired and checked by reading, not compiled, not run on hardware
//
#include <hip/hip_runtime.h>
#include <hip/hip_bf16.h>

// ---------------------------------------------------------------------------
// MoE fused experts for MI455X (gfx1250, wave32, WMMA).
//   GEMM1 + SwiGLU:  act[T,F](bf16, in d_ws) = silu(x@w13_gate) * (x@w13_up)
//   GEMM2:           out[T,D](f32)           = act @ w2
// bf16 WMMA (v_wmma_f32_16x16x32_bf16), fp32 accumulation; fp32 operands are
// converted with native v_cvt_pk_bf16_f32 while staging through LDS.
// Tiles: 128 x (64 gate + 64 up) for GEMM1, 128 x 128 for GEMM2.
// Dims are compile-time constants so strided weight loads use immediate
// offsets. B fragments run a depth-2 rolling pipeline so WMMAs wait with
// partial s_wait_dscnt instead of draining the DS queue.
// ---------------------------------------------------------------------------

constexpr int T  = 4096;
constexpr int D  = 2048;
constexpr int F  = 1408;
constexpr int E  = 8;
constexpr int F2 = 2 * F;          // 2816

typedef __bf16 v16bf __attribute__((ext_vector_type(16)));
typedef __bf16 v8bf  __attribute__((ext_vector_type(8)));
typedef float  v8f   __attribute__((ext_vector_type(8)));

#define LDK 40   // 32 bf16 of K + 8 pad => 80 B/row, keeps 16B alignment

__device__ __forceinline__ unsigned short bf16_bits(float f) {
    union { __bf16 b; unsigned short s; } v;
    v.b = (__bf16)f;                 // native f32->bf16 convert
    return v.s;
}

// A fragment (16x32 bf16): lane<16 -> row=lane,    K = {0..7, 16..23}
//                          lane>=16 -> row=lane-16, K = {8..15, 24..31}
__device__ __forceinline__ v16bf load_a_frag(const __bf16* a, int row0, int lane) {
    const int r  = row0 + (lane & 15);
    const int kh = (lane >> 4) * 8;               // 0 or 8
    const __bf16* p = a + r * LDK + kh;
    v8bf lo = *(const v8bf*)(p);                  // K = kh .. kh+7
    v8bf hi = *(const v8bf*)(p + 16);             // K = kh+16 .. kh+23
    v16bf f;
#pragma unroll
    for (int i = 0; i < 8; ++i) { f[i] = lo[i]; f[i + 8] = hi[i]; }
    return f;
}

// B fragment (32x16 bf16), LDS holds B transposed as [n][k]:
//   lane<16 -> col=lane,    K = {0..7, 16..23} ; lane>=16 -> col=lane-16, K = {8..15, 24..31}
__device__ __forceinline__ v16bf load_b_frag(const __bf16* bT, int col0, int lane) {
    const int n  = col0 + (lane & 15);
    const int kh = (lane >> 4) * 8;               // 0 or 8
    const __bf16* p = bT + n * LDK + kh;
    v8bf lo = *(const v8bf*)(p);
    v8bf hi = *(const v8bf*)(p + 16);
    v16bf f;
#pragma unroll
    for (int i = 0; i < 8; ++i) { f[i] = lo[i]; f[i + 8] = hi[i]; }
    return f;
}

__device__ __forceinline__ v8f wmma_bf16(v16bf a, v16bf b, v8f c) {
    return __builtin_amdgcn_wmma_f32_16x16x32_bf16(
        false, a, false, b, (short)0, c, false, false);
}

// ---------------------------------------------------------------------------
// GEMM1 + SwiGLU body. MASK=false for tiles fully inside the expert range.
// ---------------------------------------------------------------------------
template <bool MASK>
__device__ __forceinline__ void gemm1_body(
    const float* __restrict__ x, const float* __restrict__ w13,
    unsigned short* __restrict__ act,
    __bf16* sA, __bf16* sBg, __bf16* sBu,
    int e, int m0, int n0, int rs, int re) {

    const int tid  = threadIdx.x;
    const int lane = tid & 31;
    const int mi   = tid >> 5;      // wave = m-strip 0..7

    v8f accG[4] = {}; v8f accU[4] = {};

    const int ar = tid >> 1, ac = (tid & 1) * 16;     // A: 16 floats/thread
    const int arow = m0 + ar;
    const bool arow_ok = !MASK || ((arow >= rs) && (arow < re));
    const int bn = tid & 63, bkg = (tid >> 6) * 8;    // B: strided reads

    const float* xp  = x + (long long)arow * D + ac;
    const float* wp  = w13 + (long long)e * D * F2 + (long long)bkg * F2 + (n0 + bn);
    __bf16* dA = sA + ar * LDK + ac;
    __bf16* dg = sBg + bn * LDK + bkg;
    __bf16* du = sBu + bn * LDK + bkg;

    for (int k0 = 0; k0 < D; k0 += 32, xp += 32, wp += 32 * F2) {
        // --- stage A (128x32 fp32 -> bf16) ---
        {
            const float4* p = (const float4*)xp;
            float4 v0 = p[0], v1 = p[1], v2 = p[2], v3 = p[3];
            if (MASK && !arow_ok) {
                v0 = make_float4(0.f, 0.f, 0.f, 0.f);
                v1 = v0; v2 = v0; v3 = v0;
            }
            dA[ 0] = (__bf16)v0.x; dA[ 1] = (__bf16)v0.y; dA[ 2] = (__bf16)v0.z; dA[ 3] = (__bf16)v0.w;
            dA[ 4] = (__bf16)v1.x; dA[ 5] = (__bf16)v1.y; dA[ 6] = (__bf16)v1.z; dA[ 7] = (__bf16)v1.w;
            dA[ 8] = (__bf16)v2.x; dA[ 9] = (__bf16)v2.y; dA[10] = (__bf16)v2.z; dA[11] = (__bf16)v2.w;
            dA[12] = (__bf16)v3.x; dA[13] = (__bf16)v3.y; dA[14] = (__bf16)v3.z; dA[15] = (__bf16)v3.w;
        }
        // --- stage B gate & up (32x64 fp32 -> bf16, transposed into [n][k]) ---
        // constexpr stride => immediate instruction offsets, one base address
#pragma unroll
        for (int i = 0; i < 8; ++i) {
            dg[i] = (__bf16)wp[i * F2];
            du[i] = (__bf16)wp[i * F2 + F];
        }
        __syncthreads();

        const v16bf af = load_a_frag(sA, mi * 16, lane);

        // depth-2 rolling pipeline over the 8 B fragments (4 gate + 4 up)
        v16bf bcur = load_b_frag(sBg, 0, lane);
#pragma unroll
        for (int s = 0; s < 8; ++s) {
            v16bf bnext = bcur;
            if (s < 7) {
                const int t = s + 1;
                bnext = (t < 4) ? load_b_frag(sBg, t * 16, lane)
                                : load_b_frag(sBu, (t - 4) * 16, lane);
            }
            if (s < 4) accG[s]     = wmma_bf16(af, bcur, accG[s]);
            else       accU[s - 4] = wmma_bf16(af, bcur, accU[s - 4]);
            bcur = bnext;
        }
        __syncthreads();
    }

    // --- epilogue: SwiGLU -> bf16 act ---
    const int half = lane >> 4, nn = lane & 15;
#pragma unroll
    for (int j = 0; j < 4; ++j) {
#pragma unroll
        for (int i = 0; i < 8; ++i) {
            const int row = m0 + mi * 16 + i + 8 * half;
            if (!MASK || (row >= rs && row < re)) {
                const float g = accG[j][i];
                const float u = accU[j][i];
                const float s = g * __builtin_amdgcn_rcpf(1.0f + __expf(-g));
                const int col = n0 + j * 16 + nn;
                act[(long long)row * F + col] = bf16_bits(s * u);
            }
        }
    }
}

__global__ __launch_bounds__(256)
void moe_gemm1_swiglu(const float* __restrict__ x,
                      const int* __restrict__ group_list,
                      const float* __restrict__ w13,
                      unsigned short* __restrict__ act) {
    const int e  = blockIdx.z;
    const int m0 = blockIdx.y * 128;
    const int n0 = blockIdx.x * 64;
    const int rs = (e == 0) ? 0 : group_list[e - 1];
    const int re = group_list[e];
    if (m0 >= re || m0 + 128 <= rs) return;     // uniform: tile not in this expert

    __shared__ __attribute__((aligned(16))) __bf16 sA [128 * LDK];  // [row][k]
    __shared__ __attribute__((aligned(16))) __bf16 sBg[ 64 * LDK];  // [n][k]
    __shared__ __attribute__((aligned(16))) __bf16 sBu[ 64 * LDK];  // [n][k]

    if (m0 >= rs && m0 + 128 <= re)   // fully interior: no row masking needed
        gemm1_body<false>(x, w13, act, sA, sBg, sBu, e, m0, n0, rs, re);
    else
        gemm1_body<true >(x, w13, act, sA, sBg, sBu, e, m0, n0, rs, re);
}

// ---------------------------------------------------------------------------
// GEMM2 body: act(bf16) @ w2 -> out(f32). Tile 128x128, K = F.
// ---------------------------------------------------------------------------
template <bool MASK>
__device__ __forceinline__ void gemm2_body(
    const unsigned short* __restrict__ act, const float* __restrict__ w2,
    float* __restrict__ out, __bf16* sA, __bf16* sB,
    int e, int m0, int n0, int rs, int re) {

    const int tid  = threadIdx.x;
    const int lane = tid & 31;
    const int mi   = tid >> 5;

    v8f acc[8] = {};

    const int ar = tid >> 1, ac = (tid & 1) * 16;
    const int arow = m0 + ar;
    const bool arow_ok = !MASK || ((arow >= rs) && (arow < re));
    const int bn = tid & 127, bkg = (tid >> 7) * 16;

    const unsigned short* ap = act + (long long)arow * F + ac;
    const float* wp = w2 + (long long)e * F * D + (long long)bkg * D + (n0 + bn);
    uint4*  dA = (uint4*)(sA + ar * LDK + ac);
    __bf16* dB = sB + bn * LDK + bkg;

    for (int k0 = 0; k0 < F; k0 += 32, ap += 32, wp += 32 * D) {
        // --- stage A: act already bf16, two 16B loads/thread ---
        {
            const uint4* p = (const uint4*)ap;
            uint4 v0 = p[0], v1 = p[1];
            if (MASK && !arow_ok) { v0 = make_uint4(0u, 0u, 0u, 0u); v1 = v0; }
            dA[0] = v0; dA[1] = v1;
        }
        // --- stage B: w2 32x128 fp32 -> bf16 transposed into [n][k] ---
#pragma unroll
        for (int i = 0; i < 16; ++i) dB[i] = (__bf16)wp[i * D];
        __syncthreads();

        const v16bf af = load_a_frag(sA, mi * 16, lane);

        // depth-2 rolling pipeline over the 8 B fragments
        v16bf bcur = load_b_frag(sB, 0, lane);
#pragma unroll
        for (int s = 0; s < 8; ++s) {
            v16bf bnext = bcur;
            if (s < 7) bnext = load_b_frag(sB, (s + 1) * 16, lane);
            acc[s] = wmma_bf16(af, bcur, acc[s]);
            bcur = bnext;
        }
        __syncthreads();
    }

    const int half = lane >> 4, nn = lane & 15;
#pragma unroll
    for (int j = 0; j < 8; ++j) {
#pragma unroll
        for (int i = 0; i < 8; ++i) {
            const int row = m0 + mi * 16 + i + 8 * half;
            if (!MASK || (row >= rs && row < re)) {
                const int col = n0 + j * 16 + nn;
                out[(long long)row * D + col] = acc[j][i];
            }
        }
    }
}

__global__ __launch_bounds__(256)
void moe_gemm2(const unsigned short* __restrict__ act,
               const int* __restrict__ group_list,
               const float* __restrict__ w2,
               float* __restrict__ out) {
    const int e  = blockIdx.z;
    const int m0 = blockIdx.y * 128;
    const int n0 = blockIdx.x * 128;
    const int rs = (e == 0) ? 0 : group_list[e - 1];
    const int re = group_list[e];
    if (m0 >= re || m0 + 128 <= rs) return;

    __shared__ __attribute__((aligned(16))) __bf16 sA[128 * LDK];   // [row][k]
    __shared__ __attribute__((aligned(16))) __bf16 sB[128 * LDK];   // [n][k]

    if (m0 >= rs && m0 + 128 <= re)
        gemm2_body<false>(act, w2, out, sA, sB, e, m0, n0, rs, re);
    else
        gemm2_body<true >(act, w2, out, sA, sB, e, m0, n0, rs, re);
}

// ---------------------------------------------------------------------------
extern "C" void kernel_launch(void* const* d_in, const int* in_sizes, int n_in,
                              void* d_out, int out_size, void* d_ws, size_t ws_size,
                              hipStream_t stream) {
    const float* x          = (const float*)d_in[0];   // [T,D] fp32
    const int*   group_list = (const int*)d_in[1];     // [E] cumulative
    const float* w13        = (const float*)d_in[2];   // [E,D,2F] fp32
    const float* w2         = (const float*)d_in[3];   // [E,F,D] fp32
    float* out = (float*)d_out;                        // [T,D] fp32
    unsigned short* act = (unsigned short*)d_ws;       // [T,F] bf16 scratch (11.5 MB)

    dim3 blk(256);
    dim3 g1(F / 64, T / 128, E);    // 22 x 32 x 8
    moe_gemm1_swiglu<<<g1, blk, 0, stream>>>(x, group_list, w13, act);

    dim3 g2(D / 128, T / 128, E);   // 16 x 32 x 8
    moe_gemm2<<<g2, blk, 0, stream>>>(act, group_list, w2, out);
}
